// ProbOhemCrossEntropy2d_28793460753068
// MI455X (gfx1250) — compile-verified, matrix operand order checked by reading, not run on hardware
//
#include <hip/hip_runtime.h>
#include <hip/hip_bf16.h>

typedef __attribute__((ext_vector_type(16))) _Float16 v16h;
typedef __attribute__((ext_vector_type(8)))  float    v8f;

// Problem constants (from reference): pred (2,16,16,256,256) f32, target (2,16,256,256)
constexpr int       N_PIX     = 2 * 16 * 256 * 256;        // 2,097,152
constexpr long long CH_STRIDE = 16LL * 256 * 256;          // d*h*w = 1,048,576 (2^20)
constexpr long long B_STRIDE  = 16LL * CH_STRIDE;          // c*d*h*w
constexpr int       MIN_KEPT_K = 100000;
constexpr int       IGNORE    = 255;
constexpr int       NB1 = 4096;   // radix level 1: float bits >> 20
constexpr int       NB2 = 4096;   // level 2: (bits >> 8) & 0xFFF
constexpr int       NB3 = 256;    // level 3: bits & 0xFF
constexpr int       LOSS_BLOCKS = 1024;

// ---------------------------------------------------------------- init
__global__ void k_init(unsigned* h1, unsigned* h2, unsigned* h3,
                       unsigned* scal, float* partials) {
    int tid = threadIdx.x;
    for (int i = tid; i < NB1; i += 256) h1[i] = 0u;
    for (int i = tid; i < NB2; i += 256) h2[i] = 0u;
    for (int i = tid; i < NB3; i += 256) h3[i] = 0u;
    for (int i = tid; i < 2 * LOSS_BLOCKS; i += 256) partials[i] = 0.0f;
    if (tid < 8) scal[tid] = 0u;
    if (tid == 0) scal[0] = (unsigned)MIN_KEPT_K;
}

// ------------------------------------------------- pass 1: softmax stream
// 4 consecutive pixels per thread; 16 channels via float4 loads (B128).
__global__ void k_softmax(const float* __restrict__ pred,
                          const int*   __restrict__ tgt,
                          float* __restrict__ prob,
                          float* __restrict__ nll) {
    int t4 = blockIdx.x * blockDim.x + threadIdx.x;
    int n0 = t4 * 4;
    if (n0 >= N_PIX) return;
    int b = n0 >> 20;                 // pixels-per-batch = 2^20
    int r = n0 & 0xFFFFF;
    const float* base = pred + (long long)b * B_STRIDE + r;

    float xs[16][4];
#pragma unroll
    for (int c = 0; c < 16; c++) {
        float4 v = *(const float4*)(base + (long long)c * CH_STRIDE);
        xs[c][0] = v.x; xs[c][1] = v.y; xs[c][2] = v.z; xs[c][3] = v.w;
    }
    int4 tg = *(const int4*)(tgt + n0);
    int tv[4] = { tg.x, tg.y, tg.z, tg.w };

#pragma unroll
    for (int j = 0; j < 4; j++) {
        int  t     = tv[j];
        bool valid = (t != IGNORE);
        int  tc    = valid ? min(max(t, 0), 15) : 0;

        float m = xs[0][j];
#pragma unroll
        for (int c = 1; c < 16; c++) m = fmaxf(m, xs[c][j]);

        float s = 0.0f, et = 0.0f, xt = 0.0f;
#pragma unroll
        for (int c = 0; c < 16; c++) {
            float x = xs[c][j];
            float e = __expf(x - m);
            s += e;
            et = (c == tc) ? e : et;
            xt = (c == tc) ? x : xt;
        }
        float p  = et / s;
        float ln = __logf(s) - (xt - m);   // -log_softmax[target]
        prob[n0 + j] = valid ? p : 1.0f;
        nll[n0 + j]  = valid ? ln : -1.0f; // sign flags invalid
    }
}

// --------------------------------- radix-select level 1: LDS histogram
__global__ void k_hist1(const float* __restrict__ prob, unsigned* __restrict__ h1) {
    __shared__ unsigned h[NB1];                     // 16 KB
    for (int i = threadIdx.x; i < NB1; i += blockDim.x) h[i] = 0u;
    __syncthreads();
    int stride = gridDim.x * blockDim.x;
    for (int n = blockIdx.x * blockDim.x + threadIdx.x; n < N_PIX; n += stride) {
        unsigned bits = __float_as_uint(prob[n]);
        unsigned bin  = bits >> 20;                 // prob >= 0 -> monotone
        if (bin > NB1 - 1) bin = NB1 - 1;
        atomicAdd(&h[bin], 1u);
    }
    __syncthreads();
    for (int i = threadIdx.x; i < NB1; i += blockDim.x)
        if (h[i]) atomicAdd(&h1[i], h[i]);
}

// ------------------------- refine passes (levels 2 and 3, global atomics)
__global__ void k_refine(const float* __restrict__ prob, unsigned* __restrict__ histOut,
                         int level, const unsigned* __restrict__ sel1p,
                         const unsigned* __restrict__ sel2p) {
    unsigned s1 = *sel1p;
    unsigned s2 = (level == 3) ? *sel2p : 0u;
    int stride = gridDim.x * blockDim.x;
    for (int n = blockIdx.x * blockDim.x + threadIdx.x; n < N_PIX; n += stride) {
        unsigned bits = __float_as_uint(prob[n]);
        if (level == 2) {
            if ((bits >> 20) == s1) atomicAdd(&histOut[(bits >> 8) & 0xFFFu], 1u);
        } else {
            if ((bits >> 8) == ((s1 << 12) | s2)) atomicAdd(&histOut[bits & 0xFFu], 1u);
        }
    }
}

// -------- single-block histogram scan: find bin containing the K-th value
__global__ void k_scan(const unsigned* __restrict__ hist, int nbins,
                       const unsigned* __restrict__ kIn,
                       unsigned* __restrict__ selOut, unsigned* __restrict__ remOut,
                       int finalLevel, float* __restrict__ thrOut,
                       const unsigned* __restrict__ sel1p,
                       const unsigned* __restrict__ sel2p) {
    __shared__ unsigned csum[256];
    unsigned K = *kIn;
    int chunk = nbins / 256;
    unsigned mysum = 0;
    for (int i = 0; i < chunk; i++) mysum += hist[threadIdx.x * chunk + i];
    csum[threadIdx.x] = mysum;
    __syncthreads();
    if (threadIdx.x == 0) {
        unsigned cum = 0;
        int c = 0;
        for (; c < 256; c++) {
            if (cum + csum[c] >= K) break;
            cum += csum[c];
        }
        if (c == 256) c = 255;
        int bin = c * chunk;
        for (int i = 0; i < chunk; i++) {
            unsigned hv = hist[c * chunk + i];
            if (cum + hv >= K) { bin = c * chunk + i; break; }
            cum += hv;
        }
        *selOut = (unsigned)bin;
        *remOut = K - cum;                  // 1..hist[bin]
        if (finalLevel) {
            unsigned kbits = ((*sel1p) << 20) | ((*sel2p) << 8) | (unsigned)bin;
            *thrOut = fmaxf(0.7f, __uint_as_float(kbits));
        }
    }
}

// ------------- pass 3: masked NLL reduction; WMMA wave-reduce for losses
__global__ void k_loss(const float* __restrict__ prob, const float* __restrict__ nll,
                       const float* __restrict__ thrP, float* __restrict__ partials) {
    float thr = *thrP;
    float lsum = 0.0f;
    int   cnt  = 0;
    int stride = gridDim.x * blockDim.x;
    for (int n = blockIdx.x * blockDim.x + threadIdx.x; n < N_PIX; n += stride) {
        float p = prob[n];
        float l = nll[n];
        bool fv = (l >= 0.0f) && (p <= thr);
        if (fv) { lsum += l; cnt++; }
    }

    // ---- wave-level WMMA reduction of lsum (A = ones, B holds hi/lo f16
    //      split of each lane's partial -> near-f32 accuracy; f32 accum).
    //      D column n = lane n + lane n+16 partial sums.
    v16h ones;
#pragma unroll
    for (int i = 0; i < 16; i++) ones[i] = (_Float16)1.0f;
    v16h bm;
#pragma unroll
    for (int i = 0; i < 16; i++) bm[i] = (_Float16)0.0f;
    _Float16 hi = (_Float16)lsum;
    float    lo = lsum - (float)hi;
    bm[0] = hi;
    bm[1] = (_Float16)lo;
    v8f acc;
#pragma unroll
    for (int i = 0; i < 8; i++) acc[i] = 0.0f;
    acc = __builtin_amdgcn_wmma_f32_16x16x32_f16(false, ones, false, bm,
                                                 (short)0, acc, false, false);
    float w = acc[0];                         // colsum for column lane%16
#pragma unroll
    for (int off = 1; off < 16; off <<= 1)    // merge the 16 distinct columns
        w += __shfl_xor(w, off, 32);
#pragma unroll
    for (int off = 1; off < 32; off <<= 1)    // integer count: plain shuffles
        cnt += __shfl_xor(cnt, off, 32);

    __shared__ float wl[8];
    __shared__ int   wc[8];
    int wave = threadIdx.x >> 5;
    if ((threadIdx.x & 31) == 0) { wl[wave] = w; wc[wave] = cnt; }
    __syncthreads();
    if (threadIdx.x == 0) {
        float L = 0.0f; int C = 0;
        for (int i = 0; i < 8; i++) { L += wl[i]; C += wc[i]; }
        partials[2 * blockIdx.x]     = L;
        partials[2 * blockIdx.x + 1] = (float)C;
    }
}

// --------------------------------- final: deterministic fixed-order sum
__global__ void k_final(const float* __restrict__ partials, int nb, float* __restrict__ out) {
    __shared__ float sl[256];
    __shared__ float sc[256];
    float L = 0.0f, C = 0.0f;
    for (int i = threadIdx.x; i < nb; i += 256) {
        L += partials[2 * i];
        C += partials[2 * i + 1];
    }
    sl[threadIdx.x] = L; sc[threadIdx.x] = C;
    __syncthreads();
    for (int s = 128; s > 0; s >>= 1) {
        if (threadIdx.x < s) { sl[threadIdx.x] += sl[threadIdx.x + s];
                               sc[threadIdx.x] += sc[threadIdx.x + s]; }
        __syncthreads();
    }
    if (threadIdx.x == 0) out[0] = sl[0] / fmaxf(sc[0], 1.0f);
}

extern "C" void kernel_launch(void* const* d_in, const int* in_sizes, int n_in,
                              void* d_out, int out_size, void* d_ws, size_t ws_size,
                              hipStream_t stream) {
    const float* pred = (const float*)d_in[0];
    const int*   tgt  = (const int*)d_in[1];
    float*       out  = (float*)d_out;

    char* ws = (char*)d_ws;
    float*    prob     = (float*)ws;
    float*    nll      = (float*)(ws + (size_t)N_PIX * 4);
    unsigned* h1       = (unsigned*)(ws + (size_t)N_PIX * 8);
    unsigned* h2       = h1 + NB1;
    unsigned* h3       = h2 + NB2;
    unsigned* scal     = h3 + NB3;      // [0]=K, [1]=sel1, [2]=rem1, [3]=sel2,
                                        // [4]=rem2, [5]=sel3, [6]=rem3, [7]=thr
    float*    partials = (float*)(scal + 8);

    k_init<<<1, 256, 0, stream>>>(h1, h2, h3, scal, partials);
    k_softmax<<<(N_PIX / 4 + 255) / 256, 256, 0, stream>>>(pred, tgt, prob, nll);
    k_hist1<<<512, 256, 0, stream>>>(prob, h1);
    k_scan<<<1, 256, 0, stream>>>(h1, NB1, scal + 0, scal + 1, scal + 2,
                                  0, nullptr, scal + 1, scal + 3);
    k_refine<<<512, 256, 0, stream>>>(prob, h2, 2, scal + 1, scal + 3);
    k_scan<<<1, 256, 0, stream>>>(h2, NB2, scal + 2, scal + 3, scal + 4,
                                  0, nullptr, scal + 1, scal + 3);
    k_refine<<<512, 256, 0, stream>>>(prob, h3, 3, scal + 1, scal + 3);
    k_scan<<<1, 256, 0, stream>>>(h3, NB3, scal + 4, scal + 5, scal + 6,
                                  1, (float*)(scal + 7), scal + 1, scal + 3);
    k_loss<<<LOSS_BLOCKS, 256, 0, stream>>>(prob, nll, (const float*)(scal + 7), partials);
    k_final<<<1, 256, 0, stream>>>(partials, LOSS_BLOCKS, out);
}
